// SwinTransformerBlock_9543417332053
// MI455X (gfx1250) — compile-verified, matrix operand order checked by reading, not run on hardware
//
#include <hip/hip_runtime.h>
#include <hip/hip_bf16.h>
#include <stdint.h>

// ---------------------------------------------------------------------------
// Swin block for MI455X (gfx1250): bf16 WMMA GEMMs + fused epilogues.
// Wave32 everywhere; v_wmma_f32_16x16x32_bf16 is the matrix workhorse.
// B-operand tiles are staged into LDS with gfx1250 async global->LDS DMA
// (ASYNCcnt / s_wait_asynccnt), double-buffered against the WMMA stream.
// ---------------------------------------------------------------------------

typedef __bf16 bf16_t;
typedef bf16_t v16bf __attribute__((ext_vector_type(16)));
typedef float  v8f   __attribute__((ext_vector_type(8)));
typedef unsigned short u16;

#define HH 56
#define WW_ 56
#define WS 7
#define SHIFT 3
#define NWIN 64           // (56/7)^2
#define NTOK 49           // 7*7
#define BATCH 16
#define CDIM 512
#define NHEAD 16
#define HDIM 32
#define LSEQ (HH*WW_)     // 3136
#define MROWS (BATCH*NWIN*NTOK)   // 50176
#define N3 (3*CDIM)       // 1536
#define DH (4*CDIM)       // 2048

union Frag {
    v16bf v;
    uint4 u[2];
};

__device__ __forceinline__ u16 f2bf(float f) {
    unsigned int x = __float_as_uint(f);
    unsigned int r = (x + 0x7FFFu + ((x >> 16) & 1u)) >> 16;  // RNE
    return (u16)r;
}

// async global -> LDS, 16 bytes per lane; tracked by ASYNCcnt
__device__ __forceinline__ void async_g2l_b128(unsigned int lds_off, const void* gaddr) {
    asm volatile("global_load_async_to_lds_b128 %0, %1, off"
                 :
                 : "v"(lds_off), "v"((unsigned long long)(uintptr_t)gaddr)
                 : "memory");
}
__device__ __forceinline__ void wait_asynccnt0() {
    asm volatile("s_wait_asynccnt 0x0" ::: "memory");
}

// ---------------------------------------------------------------------------
// f32 -> bf16 elementwise (weights)
// ---------------------------------------------------------------------------
__global__ void cvt_bf16_kernel(const float* __restrict__ in, u16* __restrict__ out, int n) {
    int i = blockIdx.x * blockDim.x + threadIdx.x;
    if (i < n) out[i] = f2bf(in[i]);
}

// ---------------------------------------------------------------------------
// LayerNorm. PARTITION=1: fused roll(-3,-3) + window partition, writes the
// window-ordered bf16 activation matrix (M x 512). PARTITION=0: row-major.
// One wave32 per token; 16 elements per lane; shfl_xor reduction.
// ---------------------------------------------------------------------------
template <int PARTITION>
__global__ __launch_bounds__(128) void ln_kernel(const float* __restrict__ x,
                                                 const float* __restrict__ g,
                                                 const float* __restrict__ bp,
                                                 u16* __restrict__ out) {
    int wave = threadIdx.x >> 5;
    int lane = threadIdx.x & 31;
    int t = blockIdx.x * 4 + wave;          // token id in [0, 50176)
    const float* row = x + (size_t)t * CDIM;

    float vals[16];
    float sum = 0.f, sq = 0.f;
#pragma unroll
    for (int e = 0; e < 16; ++e) {
        float f = row[lane + e * 32];
        vals[e] = f;
        sum += f;
        sq += f * f;
    }
#pragma unroll
    for (int off = 16; off > 0; off >>= 1) {
        sum += __shfl_xor(sum, off, 32);
        sq  += __shfl_xor(sq,  off, 32);
    }
    float mean = sum * (1.f / CDIM);
    float var  = sq * (1.f / CDIM) - mean * mean;
    float rstd = rsqrtf(var + 1e-5f);

    size_t obase;
    if (PARTITION) {
        int b = t / LSEQ, l = t - b * LSEQ;
        int h = l / WW_, w = l - h * WW_;
        int hs = h + (HH - SHIFT);  if (hs >= HH) hs -= HH;   // rolled coords
        int ws = w + (WW_ - SHIFT); if (ws >= WW_) ws -= WW_;
        int wy = hs / WS, iy = hs - wy * WS;
        int wx = ws / WS, ix = ws - wx * WS;
        obase = (((size_t)b * NWIN + wy * 8 + wx) * NTOK + iy * WS + ix) * CDIM;
    } else {
        obase = (size_t)t * CDIM;
    }
#pragma unroll
    for (int e = 0; e < 16; ++e) {
        int c = lane + e * 32;
        out[obase + c] = f2bf((vals[e] - mean) * rstd * g[c] + bp[c]);
    }
}

// ---------------------------------------------------------------------------
// bf16 WMMA GEMM:  D(MxN) = A(MxK) * Bw(NxK)^T + bias, fused epilogues.
//   MODE 0: f32 store (QKV)
//   MODE 1: proj: window-reverse + unshift + residual-add into x1
//   MODE 2: exact GELU -> bf16 (MLP hidden)
//   MODE 3: residual-add with x1 -> f32 d_out (MLP out)
// Block = 256 threads = 8 waves arranged 4(M) x 2(N); wave tile 32x64;
// workgroup tile 128x128. The 128x32 B tile (8KB bf16) is staged per K-step
// into LDS with async global->LDS b128 DMA, double-buffered: one barrier per
// K-step, DMA for tile k+1 overlaps WMMA on tile k. A fragments come straight
// from global (only 2-way redundant). Fragment layout matches CDNA5 16-bit
// A/B striping: lanes 0-15 carry K[0..7]/[16..23], lanes 16-31 K[8..15]/[24..31].
// ---------------------------------------------------------------------------
template <int MODE>
__global__ __launch_bounds__(256) void gemm_bf16_kernel(
    const u16* __restrict__ A, const u16* __restrict__ Bw,
    const float* __restrict__ bias, int K, int ldn,
    float* __restrict__ outf, u16* __restrict__ outb,
    const float* __restrict__ xres) {
    __shared__ u16 Bt[2][128 * 32];   // [buffer][col_local * 32 + kk]

    int tid = threadIdx.x;
    int wave = tid >> 5, lane = tid & 31;
    int wm = wave & 3, wn = wave >> 2;
    int m0 = blockIdx.y * 128 + wm * 32;
    int nblock = blockIdx.x * 128;
    int n0 = nblock + wn * 64;
    int lr = lane & 15;
    int khalf = (lane >> 4) << 3;  // 0 or 8

    v8f zero = {0.f, 0.f, 0.f, 0.f, 0.f, 0.f, 0.f, 0.f};
    v8f acc[2][4];
#pragma unroll
    for (int i = 0; i < 2; ++i)
#pragma unroll
        for (int j = 0; j < 4; ++j) acc[i][j] = zero;

    const u16* Ar0 = A + (size_t)(m0 + lr) * K + khalf;
    const u16* Ar1 = A + (size_t)(m0 + 16 + lr) * K + khalf;

    // this thread's two 16B chunks of the 8KB B tile
    int c0 = tid, c1 = tid + 256;
    int col0 = c0 >> 2, p0 = (c0 & 3) * 8;
    int col1 = c1 >> 2, p1 = (c1 & 3) * 8;
    const u16* Bg0 = Bw + (size_t)(nblock + col0) * K + p0;
    const u16* Bg1 = Bw + (size_t)(nblock + col1) * K + p1;

    // issue first tile's DMA
    async_g2l_b128((unsigned int)(uintptr_t)&Bt[0][col0 * 32 + p0], Bg0);
    async_g2l_b128((unsigned int)(uintptr_t)&Bt[0][col1 * 32 + p1], Bg1);

    int KT = K >> 5;
    for (int kt = 0; kt < KT; ++kt) {
        int k = kt << 5;
        int buf = kt & 1;

        wait_asynccnt0();       // this wave's DMA into Bt[buf] complete
        __syncthreads();        // all waves' DMA complete; prev compute done

        if (kt + 1 < KT) {      // overlap next tile's DMA with this tile's WMMAs
            int nb = buf ^ 1, kn = k + 32;
            async_g2l_b128((unsigned int)(uintptr_t)&Bt[nb][col0 * 32 + p0], Bg0 + kn);
            async_g2l_b128((unsigned int)(uintptr_t)&Bt[nb][col1 * 32 + p1], Bg1 + kn);
        }

        Frag a0, a1, b0, b1, b2, b3;
        a0.u[0] = *(const uint4*)(Ar0 + k);
        a0.u[1] = *(const uint4*)(Ar0 + k + 16);
        a1.u[0] = *(const uint4*)(Ar1 + k);
        a1.u[1] = *(const uint4*)(Ar1 + k + 16);
        __builtin_prefetch(Ar0 + k + 32, 0, 0);
        __builtin_prefetch(Ar1 + k + 32, 0, 0);

        // B fragments from LDS (ds_load_b128 pairs on the DS pipe)
        const uint4* lb0 = (const uint4*)&Bt[buf][(wn * 64 + 0  + lr) * 32 + khalf];
        const uint4* lb1 = (const uint4*)&Bt[buf][(wn * 64 + 16 + lr) * 32 + khalf];
        const uint4* lb2 = (const uint4*)&Bt[buf][(wn * 64 + 32 + lr) * 32 + khalf];
        const uint4* lb3 = (const uint4*)&Bt[buf][(wn * 64 + 48 + lr) * 32 + khalf];
        b0.u[0] = lb0[0]; b0.u[1] = lb0[2];
        b1.u[0] = lb1[0]; b1.u[1] = lb1[2];
        b2.u[0] = lb2[0]; b2.u[1] = lb2[2];
        b3.u[0] = lb3[0]; b3.u[1] = lb3[2];

        acc[0][0] = __builtin_amdgcn_wmma_f32_16x16x32_bf16(false, a0.v, false, b0.v, (short)0, acc[0][0], false, false);
        acc[0][1] = __builtin_amdgcn_wmma_f32_16x16x32_bf16(false, a0.v, false, b1.v, (short)0, acc[0][1], false, false);
        acc[0][2] = __builtin_amdgcn_wmma_f32_16x16x32_bf16(false, a0.v, false, b2.v, (short)0, acc[0][2], false, false);
        acc[0][3] = __builtin_amdgcn_wmma_f32_16x16x32_bf16(false, a0.v, false, b3.v, (short)0, acc[0][3], false, false);
        acc[1][0] = __builtin_amdgcn_wmma_f32_16x16x32_bf16(false, a1.v, false, b0.v, (short)0, acc[1][0], false, false);
        acc[1][1] = __builtin_amdgcn_wmma_f32_16x16x32_bf16(false, a1.v, false, b1.v, (short)0, acc[1][1], false, false);
        acc[1][2] = __builtin_amdgcn_wmma_f32_16x16x32_bf16(false, a1.v, false, b2.v, (short)0, acc[1][2], false, false);
        acc[1][3] = __builtin_amdgcn_wmma_f32_16x16x32_bf16(false, a1.v, false, b3.v, (short)0, acc[1][3], false, false);
    }

    // C/D layout: lane<16 -> rows j+0..7 ; lane>=16 -> rows j+8..15 ; col = lane&15
    int rbump = (lane & 16) ? 8 : 0;
#pragma unroll
    for (int i = 0; i < 2; ++i) {
#pragma unroll
        for (int j = 0; j < 4; ++j) {
            int col = n0 + j * 16 + lr;
            float bval = bias[col];
#pragma unroll
            for (int e = 0; e < 8; ++e) {
                int row = m0 + i * 16 + rbump + e;
                float val = acc[i][j][e] + bval;
                if (MODE == 0) {
                    outf[(size_t)row * ldn + col] = val;
                } else if (MODE == 2) {
                    float gl = 0.5f * val * (1.f + erff(val * 0.70710678118654752f));
                    outb[(size_t)row * ldn + col] = f2bf(gl);
                } else if (MODE == 3) {
                    size_t idx = (size_t)row * ldn + col;
                    outf[idx] = xres[idx] + val;
                } else {  // MODE 1: window reverse + roll(+3,+3) + residual
                    int wb = row / NTOK, n = row - wb * NTOK;
                    int b = wb >> 6, w = wb & 63;
                    int wy = w >> 3, wx = w & 7;
                    int iy = n / WS, ix = n - iy * WS;
                    int hs = wy * WS + iy, ws = wx * WS + ix;
                    int h = hs + SHIFT; if (h >= HH) h -= HH;
                    int wc = ws + SHIFT; if (wc >= WW_) wc -= WW_;
                    size_t idx = (((size_t)b * LSEQ) + h * WW_ + wc) * CDIM + col;
                    outf[idx] = xres[idx] + val;
                }
            }
        }
    }
}

// ---------------------------------------------------------------------------
// Per-(window, head) attention: scores + rel-pos-bias + shifted-window mask +
// softmax + S@V, all in LDS. Writes bf16 output laid out as proj's A operand.
// ---------------------------------------------------------------------------
__global__ __launch_bounds__(128) void attn_kernel(const float* __restrict__ qkv,
                                                   const int* __restrict__ rpi,
                                                   const float* __restrict__ mask,
                                                   const float* __restrict__ rpb_table,
                                                   u16* __restrict__ obf) {
    __shared__ float q[NTOK * HDIM];
    __shared__ float kk[NTOK * HDIM];
    __shared__ float vv[NTOK * HDIM];
    __shared__ float S[NTOK * NTOK];

    int wb = blockIdx.x;    // window-batch [0, 1024)
    int head = blockIdx.y;  // [0, 16)
    int tid = threadIdx.x;
    const float scale = 0.1767766952966369f;  // 32^-0.5

    size_t base = (size_t)wb * NTOK * N3 + head * HDIM;
    for (int idx = tid; idx < NTOK * HDIM; idx += 128) {
        int n = idx >> 5, d = idx & 31;
        size_t r = base + (size_t)n * N3 + d;
        q[idx]  = qkv[r] * scale;
        kk[idx] = qkv[r + CDIM];
        vv[idx] = qkv[r + 2 * CDIM];
    }
    __syncthreads();

    int w = wb & (NWIN - 1);
    const float* mrow = mask + (size_t)w * NTOK * NTOK;
    for (int idx = tid; idx < NTOK * NTOK; idx += 128) {
        int i = idx / NTOK, j = idx - i * NTOK;
        float s = 0.f;
#pragma unroll
        for (int d = 0; d < HDIM; ++d) s += q[i * HDIM + d] * kk[j * HDIM + d];
        s += rpb_table[rpi[idx] * NHEAD + head];
        s += mrow[idx];
        S[idx] = s;
    }
    __syncthreads();

    for (int i = tid; i < NTOK; i += 128) {
        float mx = -1e30f;
        for (int j = 0; j < NTOK; ++j) mx = fmaxf(mx, S[i * NTOK + j]);
        float sum = 0.f;
        for (int j = 0; j < NTOK; ++j) {
            float e = __expf(S[i * NTOK + j] - mx);
            S[i * NTOK + j] = e;
            sum += e;
        }
        float inv = 1.f / sum;
        for (int j = 0; j < NTOK; ++j) S[i * NTOK + j] *= inv;
    }
    __syncthreads();

    for (int idx = tid; idx < NTOK * HDIM; idx += 128) {
        int i = idx >> 5, d = idx & 31;
        float acc = 0.f;
#pragma unroll 7
        for (int j = 0; j < NTOK; ++j) acc += S[i * NTOK + j] * vv[j * HDIM + d];
        obf[((size_t)wb * NTOK + i) * CDIM + head * HDIM + d] = f2bf(acc);
    }
}

// ---------------------------------------------------------------------------
// Host-side launch
// ---------------------------------------------------------------------------
extern "C" void kernel_launch(void* const* d_in, const int* in_sizes, int n_in,
                              void* d_out, int out_size, void* d_ws, size_t ws_size,
                              hipStream_t stream) {
    const float* x        = (const float*)d_in[0];
    const int*   rpi      = (const int*)d_in[1];
    const float* amask    = (const float*)d_in[2];
    const float* n1g      = (const float*)d_in[3];
    const float* n1b      = (const float*)d_in[4];
    const float* qkv_w    = (const float*)d_in[5];
    const float* qkv_b    = (const float*)d_in[6];
    const float* proj_w   = (const float*)d_in[7];
    const float* proj_b   = (const float*)d_in[8];
    const float* rpb_tab  = (const float*)d_in[9];
    const float* n2g      = (const float*)d_in[10];
    const float* n2b      = (const float*)d_in[11];
    const float* mlp_w1   = (const float*)d_in[12];
    const float* mlp_b1   = (const float*)d_in[13];
    const float* mlp_w2   = (const float*)d_in[14];
    const float* mlp_b2   = (const float*)d_in[15];
    float* out = (float*)d_out;

    char* ws = (char*)d_ws;
    size_t off = 0;
    auto alloc = [&](size_t bytes) { size_t o = off; off += (bytes + 255) & ~(size_t)255; return o; };

    size_t o_xw  = alloc((size_t)MROWS * CDIM * 2);   // LN1 output (bf16); reused as h2
    size_t o_qkv = alloc((size_t)MROWS * N3 * 4);     // qkv f32; reused as MLP hidden bf16
    size_t o_att = alloc((size_t)MROWS * CDIM * 2);   // attention out bf16
    size_t o_x1  = alloc((size_t)MROWS * CDIM * 4);   // post-attn residual f32
    size_t o_wq  = alloc((size_t)N3 * CDIM * 2);
    size_t o_wp  = alloc((size_t)CDIM * CDIM * 2);
    size_t o_w1  = alloc((size_t)DH * CDIM * 2);
    size_t o_w2  = alloc((size_t)CDIM * DH * 2);

    u16* xw_bf = (u16*)(ws + o_xw);
    float* qkv = (float*)(ws + o_qkv);
    u16* m_bf  = (u16*)(ws + o_qkv);  // reuse
    u16* h2_bf = (u16*)(ws + o_xw);   // reuse
    u16* at_bf = (u16*)(ws + o_att);
    float* x1  = (float*)(ws + o_x1);
    u16* wq_bf = (u16*)(ws + o_wq);
    u16* wp_bf = (u16*)(ws + o_wp);
    u16* w1_bf = (u16*)(ws + o_w1);
    u16* w2_bf = (u16*)(ws + o_w2);

    // 1. weight conversion to bf16
    cvt_bf16_kernel<<<(N3 * CDIM + 255) / 256, 256, 0, stream>>>(qkv_w, wq_bf, N3 * CDIM);
    cvt_bf16_kernel<<<(CDIM * CDIM + 255) / 256, 256, 0, stream>>>(proj_w, wp_bf, CDIM * CDIM);
    cvt_bf16_kernel<<<(DH * CDIM + 255) / 256, 256, 0, stream>>>(mlp_w1, w1_bf, DH * CDIM);
    cvt_bf16_kernel<<<(CDIM * DH + 255) / 256, 256, 0, stream>>>(mlp_w2, w2_bf, CDIM * DH);

    // 2. LN1 + shift + window partition -> bf16 A operand
    ln_kernel<1><<<MROWS / 4, 128, 0, stream>>>(x, n1g, n1b, xw_bf);

    // 3. QKV GEMM (50176 x 512 x 1536)
    gemm_bf16_kernel<0><<<dim3(N3 / 128, MROWS / 128), 256, 0, stream>>>(
        xw_bf, wq_bf, qkv_b, CDIM, N3, qkv, nullptr, nullptr);

    // 4. windowed attention
    attn_kernel<<<dim3(BATCH * NWIN, NHEAD), 128, 0, stream>>>(qkv, rpi, amask, rpb_tab, at_bf);

    // 5. proj GEMM + window reverse + unshift + residual -> x1
    gemm_bf16_kernel<1><<<dim3(CDIM / 128, MROWS / 128), 256, 0, stream>>>(
        at_bf, wp_bf, proj_b, CDIM, CDIM, x1, nullptr, x);

    // 6. LN2 -> bf16
    ln_kernel<0><<<MROWS / 4, 128, 0, stream>>>(x1, n2g, n2b, h2_bf);

    // 7. MLP1 GEMM + GELU -> bf16 hidden
    gemm_bf16_kernel<2><<<dim3(DH / 128, MROWS / 128), 256, 0, stream>>>(
        h2_bf, w1_bf, mlp_b1, CDIM, DH, nullptr, m_bf, nullptr);

    // 8. MLP2 GEMM + residual -> d_out
    gemm_bf16_kernel<3><<<dim3(CDIM / 128, MROWS / 128), 256, 0, stream>>>(
        m_bf, w2_bf, mlp_b2, DH, CDIM, out, nullptr, x1);
}
